// SingleHeadAttentionWithKVCache_30932354465931
// MI455X (gfx1250) — matrix operand order repaired and checked by reading, not compiled, not run on hardware
//
#include <hip/hip_runtime.h>
#include <cstdint>

// ---------------- problem constants ----------------
#define BB   32      // batch
#define TQ   8       // query length
#define TCP  4096    // past cache length
#define TCC  4104    // total cache length (TCP + TQ)
#define DD   1024    // model dim
#define CH   512     // keys per attention chunk (multiple of 32)
#define NCHUNK 9     // ceil(4104 / 512)
#define TDM_ELEMS 16384  // elements per TDM staging tile (64 KB fp32)

typedef __attribute__((ext_vector_type(16))) __bf16 v16bf;
typedef __attribute__((ext_vector_type(8)))  float  v8f;
typedef uint32_t u32x4 __attribute__((ext_vector_type(4)));
typedef uint32_t u32x8 __attribute__((ext_vector_type(8)));

__device__ __forceinline__ int imin(int a, int b) { return a < b ? a : b; }

__device__ __forceinline__ v8f wmma_bf16(v16bf a, v16bf b, v8f c) {
  return __builtin_amdgcn_wmma_f32_16x16x32_bf16(false, a, false, b, (short)0, c,
                                                 false, false);
}

// A-matrix 16x32 bf16 tile from row-major fp32 (row stride lda).
// Lane l (r = l&15, h = l>>4): elems 0..15 -> K = 16h + j
__device__ __forceinline__ v16bf load_a_f32(const float* __restrict__ base, int lda,
                                            int m0, int k0, int lane) {
  const int r = lane & 15, h = lane >> 4;
  const float* p = base + (size_t)(m0 + r) * lda + k0 + 16 * h;
  v16bf a;
#pragma unroll
  for (int j = 0; j < 8; ++j) a[j] = (__bf16)p[j];
#pragma unroll
  for (int j = 0; j < 8; ++j) a[8 + j] = (__bf16)p[8 + j];
  return a;
}

// B-matrix 32x16 bf16 tile where this lane's column is a contiguous fp32 row:
// lane holds K = 16h + j, j = 0..15 (16 consecutive floats at rowk0).
__device__ __forceinline__ v16bf load_b_row(const float* __restrict__ rowk0, int h) {
  const float* p = rowk0 + 16 * h;
  v16bf b;
#pragma unroll
  for (int j = 0; j < 16; ++j) b[j] = (__bf16)p[j];
  return b;
}

// ---------------------------------------------------------------------------
// TDM 1-D async copy (global -> LDS -> global), nelem fp32 elements.
// D# packing per CDNA5 ISA 8.3/8.4. Same-wave tensor ops execute in order, so
// a single staging buffer is reused across load/store pairs; completion is
// tracked by TENSORcnt (explicit s_wait_tensorcnt + implicit wait at ENDPGM).
// ---------------------------------------------------------------------------
__device__ __forceinline__ void tdm_copy_1d(uint32_t lds_addr, uint64_t src,
                                            uint64_t dst, uint32_t nelem) {
  u32x8 g1;
  g1[0] = 2u << 16;                       // workgroup_mask=0, data_size=4B
  g1[1] = (nelem & 0xFFFFu) << 16;        // tensor_dim0[15:0]
  g1[2] = ((nelem >> 16) & 0xFFFFu) | (1u << 16);  // tensor_dim0[31:16], tensor_dim1=1
  g1[3] = (nelem & 0xFFFFu) << 16;        // tensor_dim1[31:16]=0, tile_dim0=nelem
  g1[4] = 1u;                             // tile_dim1=1, tile_dim2=0
  g1[5] = nelem;                          // tensor_dim0_stride[31:0]
  g1[6] = 0u;                             // stride msbs, tensor_dim1_stride lo
  g1[7] = 0u;

  u32x4 g0;
  g0[0] = 1u;                             // count=1, no gather
  g0[1] = lds_addr;                       // LDS byte address
  g0[2] = (uint32_t)src;                  // global_addr[31:0]
  g0[3] = ((uint32_t)(src >> 32) & 0x1FFFFFFu) | (2u << 30);  // addr[56:32], type=2
  asm volatile("tensor_load_to_lds %0, %1" ::"s"(g0), "s"(g1) : "memory");

  g0[2] = (uint32_t)dst;
  g0[3] = ((uint32_t)(dst >> 32) & 0x1FFFFFFu) | (2u << 30);
  asm volatile("tensor_store_from_lds %0, %1" ::"s"(g0), "s"(g1) : "memory");
}

// ---------------------------------------------------------------------------
// Projection GEMM: dst = A[256,1024] @ W^T + bias  (torch linear convention)
// grid (8,16), block 256 (8 waves). wave -> one 16x16 output tile column group.
// mode 0: dst[row*D + e]                  (final output)
// mode 1: dst[(b*16 + t)*D + e]           (Q workspace, padded to 16 rows)
// mode 2: dst[(b*TCC + TCP + t)*D + e]    (k_new / v_new tail of cache output)
// ---------------------------------------------------------------------------
__global__ void proj_gemm_kernel(const float* __restrict__ A,
                                 const float* __restrict__ W,
                                 const float* __restrict__ bias,
                                 float* __restrict__ dst, int mode) {
  const int lane = threadIdx.x & 31;
  const int wave = threadIdx.x >> 5;
  const int r16 = lane & 15, h = lane >> 4;
  const int e0 = (blockIdx.x * 8 + wave) * 16;  // output-feature tile
  const int m0 = blockIdx.y * 16;               // row tile

  const float* wrow = W + (size_t)(e0 + r16) * DD;  // this lane's B column = W row
  v8f acc = {};
#pragma unroll 4
  for (int kk = 0; kk < DD; kk += 32) {
    v16bf a = load_a_f32(A, DD, m0, kk, lane);
    v16bf b = load_b_row(wrow + kk, h);
    acc = wmma_bf16(a, b, acc);
  }
  const int e = e0 + r16;
  const float bz = bias[e];
#pragma unroll
  for (int i = 0; i < 8; ++i) {
    const int m = m0 + 8 * h + i;   // global row in [0,256)
    const int b = m >> 3, t = m & 7;
    size_t off;
    if (mode == 0)      off = (size_t)m * DD + e;
    else if (mode == 1) off = ((size_t)b * 16 + t) * DD + e;
    else                off = ((size_t)b * TCC + TCP + t) * DD + e;
    dst[off] = acc[i] + bz;
  }
}

// zero helper (Q workspace pad rows must be 0)
__global__ void zero_f4_kernel(float4* __restrict__ p, unsigned n4) {
  unsigned i = blockIdx.x * 256u + threadIdx.x;
  if (i < n4) p[i] = make_float4(0.f, 0.f, 0.f, 0.f);
}

// ---------------------------------------------------------------------------
// Fused: TDM copy past_k/past_v chunk -> output k/v (async, overlaps compute)
//        + flash-attention partial (split-K over key chunks).
// grid (NCHUNK, B), block 256 (8 waves).
// Produces per-(b,chunk): m[16], l[16], ctx_partial[16,1024].
// ---------------------------------------------------------------------------
__global__ void attn_partial_kernel(const float* __restrict__ past_k,
                                    const float* __restrict__ past_v,
                                    const float* __restrict__ qws,
                                    float* __restrict__ kout,
                                    float* __restrict__ vout,
                                    float* __restrict__ ctxp,
                                    float* __restrict__ mlb) {
  const int c = blockIdx.x;     // key chunk
  const int b = blockIdx.y;     // batch
  const int s0 = c * CH;
  const int cnt = imin(CH, TCC - s0);
  const int NT = (cnt + 15) >> 4;          // 16-wide score tiles
  const int tid = threadIdx.x;
  const int lane = tid & 31;
  const int wave = tid >> 5;
  const int r16 = lane & 15, h = lane >> 4;

  __shared__ __align__(16) float  sS[16][CH];     // raw scores (32 KB)
  __shared__ __align__(16) __bf16 sP[16][CH];     // exp(s - m) probs (16 KB)
  __shared__ __align__(16) float  stage[TDM_ELEMS];  // TDM staging (64 KB)
  __shared__ float sM[16], sL[16];

  // ---- phase 0: zero sP; kick off async TDM copy of past rows to k/v out ----
  {
    uint4 z = make_uint4(0u, 0u, 0u, 0u);
    for (int i = tid; i < (16 * CH * 2) / 16; i += 256) ((uint4*)sP)[i] = z;
  }
  if (wave == 0) {
    const int cend = imin(s0 + cnt, TCP);
    if (cend > s0) {
      const uint32_t lds = (uint32_t)(uintptr_t)&stage[0];
      const size_t nelems = (size_t)(cend - s0) * DD;   // multiple of TDM_ELEMS
      const uint64_t srck = (uint64_t)(uintptr_t)(past_k + ((size_t)b * TCP + s0) * DD);
      const uint64_t srcv = (uint64_t)(uintptr_t)(past_v + ((size_t)b * TCP + s0) * DD);
      const uint64_t dstk = (uint64_t)(uintptr_t)(kout + ((size_t)b * TCC + s0) * DD);
      const uint64_t dstv = (uint64_t)(uintptr_t)(vout + ((size_t)b * TCC + s0) * DD);
      for (size_t e0 = 0; e0 < nelems; e0 += TDM_ELEMS) {
        tdm_copy_1d(lds, srck + e0 * 4, dstk + e0 * 4, TDM_ELEMS);
        tdm_copy_1d(lds, srcv + e0 * 4, dstv + e0 * 4, TDM_ELEMS);
      }
    }
  }

  // ---- phase 1: scores S[16, cnt] = Q_b @ K_chunk^T * scale (WMMA bf16) ----
  const float* qbase = qws + (size_t)b * 16 * DD;
  for (int j = wave; j < NT; j += 8) {
    const int scol = s0 + j * 16 + r16;              // this lane's key index
    const int scl = imin(scol, TCC - 1);
    const float* krow = (scl < TCP)
                            ? past_k + ((size_t)b * TCP + scl) * DD
                            : kout + ((size_t)b * TCC + scl) * DD;
    v8f acc = {};
#pragma unroll 4
    for (int kk = 0; kk < DD; kk += 32) {
      v16bf a = load_a_f32(qbase, DD, 0, kk, lane);
      v16bf bm = load_b_row(krow + kk, h);
      acc = wmma_bf16(a, bm, acc);
    }
    const float scale = 0.03125f;  // 1/sqrt(1024)
    const int col = j * 16 + r16;
    const bool valid = (scol < TCC);
#pragma unroll
    for (int i = 0; i < 8; ++i)
      sS[8 * h + i][col] = valid ? acc[i] * scale : -__builtin_inff();
  }
  __syncthreads();

  // ---- phase 2: per-row max / exp / sum; stage probs as bf16 ----
  {
    const int r = tid >> 4, c0 = tid & 15;  // 16 threads per row
    const int ncols = NT * 16;
    float m = -__builtin_inff();
    for (int jj = c0; jj < ncols; jj += 16) m = fmaxf(m, sS[r][jj]);
#pragma unroll
    for (int off = 8; off; off >>= 1) m = fmaxf(m, __shfl_xor(m, off, 16));
    float ssum = 0.f;
    for (int jj = c0; jj < ncols; jj += 16) {
      float e = __expf(sS[r][jj] - m);
      sP[r][jj] = (__bf16)e;
      ssum += e;
    }
#pragma unroll
    for (int off = 8; off; off >>= 1) ssum += __shfl_xor(ssum, off, 16);
    if (c0 == 0) {
      sM[r] = m;
      sL[r] = ssum;
      mlb[(((size_t)b * NCHUNK + c) * 16 + r) * 2 + 0] = m;
      mlb[(((size_t)b * NCHUNK + c) * 16 + r) * 2 + 1] = ssum;
    }
  }
  __syncthreads();

  // ---- phase 3: ctx_partial[16, 1024] = P @ V_chunk (WMMA bf16) ----
  const int NTK = (NT * 16 + 31) >> 5;  // K steps of 32 over probs
  const int d0w = wave * 128;           // each wave owns 128 d-columns
  float* dstp = ctxp + ((size_t)b * NCHUNK + c) * 16 * DD;
#pragma unroll
  for (int nt = 0; nt < 8; ++nt) {
    const int dc = d0w + nt * 16 + r16;  // this lane's d column
    v8f acc = {};
    for (int ks = 0; ks < NTK; ++ks) {
      const int k0 = ks * 32;
      v16bf a;
#pragma unroll
      for (int j2 = 0; j2 < 8; ++j2) a[j2] = sP[r16][k0 + 16 * h + j2];
#pragma unroll
      for (int j2 = 0; j2 < 8; ++j2) a[8 + j2] = sP[r16][k0 + 16 * h + 8 + j2];
      v16bf bm;
#pragma unroll
      for (int j2 = 0; j2 < 16; ++j2) {
        int sg = s0 + k0 + 16 * h + j2;
        int sc = imin(sg, TCC - 1);       // clamped rows multiply prob 0
        const float* vrow = (sc < TCP)
                                ? past_v + ((size_t)b * TCP + sc) * DD
                                : vout + ((size_t)b * TCC + sc) * DD;
        bm[j2] = (__bf16)vrow[dc];
      }
      acc = wmma_bf16(a, bm, acc);
    }
#pragma unroll
    for (int i = 0; i < 8; ++i) dstp[(size_t)(8 * h + i) * DD + dc] = acc[i];
  }

  // Drain outstanding TDM copies (ENDPGM also waits, this makes it explicit).
  if (wave == 0) __builtin_amdgcn_s_wait_tensorcnt(0);
}

// ---------------------------------------------------------------------------
// Combine split-K partials: ctx[b*8+t, :] = sum_c w_c * ctxp_c / L
// ---------------------------------------------------------------------------
__global__ void combine_kernel(const float* __restrict__ ctxp,
                               const float* __restrict__ mlb,
                               float* __restrict__ ctx) {
  const int b = blockIdx.x;
  const int tid = threadIdx.x;
  __shared__ float w[16][NCHUNK];
  __shared__ float invL[16];
  if (tid < 16) {
    float m = -__builtin_inff();
    for (int c = 0; c < NCHUNK; ++c)
      m = fmaxf(m, mlb[(((size_t)b * NCHUNK + c) * 16 + tid) * 2]);
    float L = 0.f;
    for (int c = 0; c < NCHUNK; ++c) {
      float mc = mlb[(((size_t)b * NCHUNK + c) * 16 + tid) * 2 + 0];
      float lc = mlb[(((size_t)b * NCHUNK + c) * 16 + tid) * 2 + 1];
      float e = __expf(mc - m);
      w[tid][c] = e;
      L += e * lc;
    }
    invL[tid] = 1.f / L;
  }
  __syncthreads();
  for (int t = 0; t < TQ; ++t) {
    const float il = invL[t];
    for (int d = tid; d < DD; d += 256) {
      float acc = 0.f;
#pragma unroll
      for (int c = 0; c < NCHUNK; ++c)
        acc += w[t][c] *
               ctxp[(((size_t)b * NCHUNK + c) * 16 + t) * (size_t)DD + d];
      ctx[((size_t)b * TQ + t) * DD + d] = acc * il;
    }
  }
}

// ---------------------------------------------------------------------------
extern "C" void kernel_launch(void* const* d_in, const int* in_sizes, int n_in,
                              void* d_out, int out_size, void* d_ws, size_t ws_size,
                              hipStream_t stream) {
  const float* x      = (const float*)d_in[0];
  const float* past_k = (const float*)d_in[1];
  const float* past_v = (const float*)d_in[2];
  const float* Wq = (const float*)d_in[3];
  const float* bq = (const float*)d_in[4];
  const float* Wk = (const float*)d_in[5];
  const float* bk = (const float*)d_in[6];
  const float* Wv = (const float*)d_in[7];
  const float* bv = (const float*)d_in[8];
  const float* Wo = (const float*)d_in[9];
  const float* bo = (const float*)d_in[10];

  float* out  = (float*)d_out;                               // [32,8,1024]
  float* kout = out + (size_t)BB * TQ * DD;                  // [32,4104,1024]
  float* vout = kout + (size_t)BB * TCC * DD;                // [32,4104,1024]

  float* ws   = (float*)d_ws;
  float* qws  = ws;                                          // 32*16*1024
  float* ctxp = qws + (size_t)BB * 16 * DD;                  // 32*9*16*1024
  float* mlb  = ctxp + (size_t)BB * NCHUNK * 16 * DD;        // 32*9*16*2
  float* ctxc = mlb + (size_t)BB * NCHUNK * 16 * 2;          // 32*8*1024

  // 0) zero the padded Q workspace (pad rows 8..15 per batch must be 0)
  {
    unsigned n4 = (unsigned)((size_t)BB * 16 * DD / 4);
    zero_f4_kernel<<<(n4 + 255) / 256, 256, 0, stream>>>((float4*)qws, n4);
  }
  // 1) projections: Q -> workspace; k_new/v_new -> tails of cache outputs
  proj_gemm_kernel<<<dim3(8, 16), 256, 0, stream>>>(x, Wq, bq, qws, 1);
  proj_gemm_kernel<<<dim3(8, 16), 256, 0, stream>>>(x, Wk, bk, kout, 2);
  proj_gemm_kernel<<<dim3(8, 16), 256, 0, stream>>>(x, Wv, bv, vout, 2);
  // 2) fused TDM KV-cache copy + flash-attention partials (split-K)
  attn_partial_kernel<<<dim3(NCHUNK, BB), 256, 0, stream>>>(
      past_k, past_v, qws, kout, vout, ctxp, mlb);
  // 3) combine partials -> context [256,1024]
  combine_kernel<<<BB, 256, 0, stream>>>(ctxp, mlb, ctxc);
  // 4) output projection
  proj_gemm_kernel<<<dim3(8, 16), 256, 0, stream>>>(ctxc, Wo, bo, out, 0);
}
